// DoubleCrossAttentionNodesGIN_11570641895561
// MI455X (gfx1250) — matrix operand order repaired and checked by reading, not compiled, not run on hardware
//
#include <hip/hip_runtime.h>
#include <hip/hip_bf16.h>

typedef __attribute__((ext_vector_type(16))) __bf16 v16bf;
typedef __attribute__((ext_vector_type(8)))  float  v8f;

#define DEV __device__ __forceinline__

// ---------------- sizes ----------------
#define BB     32
#define NSEQ   512
#define DM     128
#define BNROWS 16384   // BB*NSEQ

static constexpr size_t F128 = (size_t)BNROWS * 128;
// float workspace offsets
static constexpr size_t OFF_AGG  = 0;
static constexpr size_t OFF_HMID = OFF_AGG  + F128;
static constexpr size_t OFF_XA1  = OFF_HMID + F128;
static constexpr size_t OFF_XA2  = OFF_XA1  + F128;
static constexpr size_t OFF_XB1  = OFF_XA2  + F128;
static constexpr size_t OFF_XB2  = OFF_XB1  + F128;
static constexpr size_t OFF_QB   = OFF_XB2  + F128;
static constexpr size_t OFF_KB   = OFF_QB   + F128;
static constexpr size_t OFF_VB   = OFF_KB   + F128;
static constexpr size_t OFF_AO   = OFF_VB   + F128;
static constexpr size_t OFF_PJ   = OFF_AO   + F128;
static constexpr size_t OFF_XL   = OFF_PJ   + F128;
static constexpr size_t OFF_FF1  = OFF_XL   + F128;              // BNROWS*256
static constexpr size_t OFF_FF2  = OFF_FF1  + (size_t)BNROWS*256;
static constexpr size_t OFF_MO   = OFF_FF2  + F128;
static constexpr size_t OFF_HA   = OFF_MO   + F128;              // 32*256
static constexpr size_t OFF_HB   = OFF_HA   + 32*256;
static constexpr size_t OFF_CQ   = OFF_HB   + 32*256;
static constexpr size_t OFF_CK   = OFF_CQ   + 32*256;
static constexpr size_t OFF_CV   = OFF_CK   + 32*256;
static constexpr size_t OFF_CAT  = OFF_CV   + 32*256;
static constexpr size_t OFF_CPJ  = OFF_CAT  + 32*256;
static constexpr size_t OFF_CXL  = OFF_CPJ  + 32*256;
static constexpr size_t OFF_CF1  = OFF_CXL  + 32*256;            // 32*512
static constexpr size_t OFF_CF2  = OFF_CF1  + 32*512;
static constexpr size_t OFF_HCAT = OFF_CF2  + 32*256;            // 32*512
static constexpr size_t OFF_FLOAT_END = OFF_HCAT + 32*512;

// packed bf16 weight offsets (elements), relative to pack base
static constexpr size_t PW_EA_C1_W1 = 0;
static constexpr size_t PW_EA_C1_W2 = PW_EA_C1_W1 + 64*128;
static constexpr size_t PW_EA_C2_W1 = PW_EA_C1_W2 + 128*128;
static constexpr size_t PW_EA_C2_W2 = PW_EA_C2_W1 + 128*128;
static constexpr size_t PW_EB_C1_W1 = PW_EA_C2_W2 + 128*128;
static constexpr size_t PW_EB_C1_W2 = PW_EB_C1_W1 + 64*128;
static constexpr size_t PW_EB_C2_W1 = PW_EB_C1_W2 + 128*128;
static constexpr size_t PW_EB_C2_W2 = PW_EB_C2_W1 + 128*128;
static constexpr size_t PW_M_WQ = PW_EB_C2_W2 + 128*128;
static constexpr size_t PW_M_WK = PW_M_WQ + 128*128;
static constexpr size_t PW_M_WV = PW_M_WK + 128*128;
static constexpr size_t PW_M_WO = PW_M_WV + 128*128;
static constexpr size_t PW_M_F1 = PW_M_WO + 128*128;
static constexpr size_t PW_M_F2 = PW_M_F1 + 128*256;
static constexpr size_t PW_C_WQ = PW_M_F2 + 256*128;
static constexpr size_t PW_C_WK = PW_C_WQ + 256*256;
static constexpr size_t PW_C_WV = PW_C_WK + 256*256;
static constexpr size_t PW_C_WO = PW_C_WV + 256*256;
static constexpr size_t PW_C_F1 = PW_C_WO + 256*256;
static constexpr size_t PW_C_F2 = PW_C_F1 + 256*512;

// ---------------- device helpers ----------------
// Native f32 -> bf16 conversion (clang lowers to v_cvt_pk_bf16_f32 on gfx1250,
// replacing the software add3/mov_b16 rounding chains).
DEV __bf16 f2bf(float f) { return (__bf16)f; }

DEV float half16_max(float v) {
  v = fmaxf(v, __shfl_xor(v, 1, 32));
  v = fmaxf(v, __shfl_xor(v, 2, 32));
  v = fmaxf(v, __shfl_xor(v, 4, 32));
  v = fmaxf(v, __shfl_xor(v, 8, 32));
  return v;
}
DEV float half16_sum(float v) {
  v += __shfl_xor(v, 1, 32);
  v += __shfl_xor(v, 2, 32);
  v += __shfl_xor(v, 4, 32);
  v += __shfl_xor(v, 8, 32);
  return v;
}
DEV float wave_sum(float v) {
  v += __shfl_xor(v, 16, 32);
  v += __shfl_xor(v, 8, 32);
  v += __shfl_xor(v, 4, 32);
  v += __shfl_xor(v, 2, 32);
  v += __shfl_xor(v, 1, 32);
  return v;
}

// ---------------- kernels ----------------
__global__ void copy_f32_kernel(const float* __restrict__ src, float* __restrict__ dst, size_t n) {
  size_t i = (size_t)blockIdx.x * blockDim.x + threadIdx.x;
  if (i < n) dst[i] = src[i];
}

// GIN edge scatter-add: Agg[dst] += X[src], 4 features per thread
__global__ void edge_scatter_kernel(const int* __restrict__ src, const int* __restrict__ dst,
                                    const float* __restrict__ X, float* __restrict__ Agg,
                                    int E, int W) {
  int idx = blockIdx.x * blockDim.x + threadIdx.x;
  int g4 = W >> 2;
  if (idx >= E * g4) return;
  int e = idx / g4;
  int c = (idx - e * g4) << 2;
  int s = src[e], d = dst[e];
  const float* xs = X + (size_t)s * W + c;
  float* ad = Agg + (size_t)d * W + c;
  atomicAdd(ad + 0, xs[0]);
  atomicAdd(ad + 1, xs[1]);
  atomicAdd(ad + 2, xs[2]);
  atomicAdd(ad + 3, xs[3]);
}

// Pack f32 weight [K,N] row-major into WMMA bf16 B-fragment order:
// Wp[((ks*ntiles+nt)*32 + lane)*16 + e] = W[(ks*32 + 16*(lane>=16) + e)*N + nt*16 + lane%16]
__global__ void pack_w_kernel(const float* __restrict__ W, __bf16* __restrict__ Wp, int K, int N) {
  int idx = blockIdx.x * blockDim.x + threadIdx.x;
  if (idx >= K * N) return;
  int e    = idx & 15;
  int lane = (idx >> 4) & 31;
  int t    = idx >> 9;
  int ntiles = N >> 4;
  int nt = t % ntiles;
  int ks = t / ntiles;
  int krow = ks * 32 + ((lane < 16) ? 0 : 16) + e;
  int col  = nt * 16 + (lane & 15);
  Wp[idx] = f2bf(W[(size_t)krow * N + col]);
}

// Generic fused GEMM: Out = act((A@W + bias) * (scale*scale_c) + shift)
// A: [M,K] f32 row-major, W packed bf16. One wave per 16x16 output tile.
__launch_bounds__(256)
__global__ void gemm_wmma_kernel(const float* __restrict__ A, const __bf16* __restrict__ Wp,
                                 const float* __restrict__ bias,
                                 const float* __restrict__ scale, const float* __restrict__ shift,
                                 float scale_c, float* __restrict__ Out,
                                 int M, int K, int N, int act) {
  int wave = (blockIdx.x * blockDim.x + threadIdx.x) >> 5;
  int lane = threadIdx.x & 31;
  int ntiles = N >> 4;
  int tw = (M >> 4) * ntiles;
  if (wave >= tw) return;
  int mt = wave / ntiles;
  int nt = wave % ntiles;
  int row = mt * 16 + (lane & 15);
  int kb  = (lane < 16) ? 0 : 16;
  v8f acc = {};
  int ksteps = K >> 5;
  for (int ks = 0; ks < ksteps; ++ks) {
    const float* ap = A + (size_t)row * K + ks * 32 + kb;
    __builtin_prefetch(ap + 32, 0, 1);
    v16bf af;
#pragma unroll
    for (int j = 0; j < 16; ++j) { af[j] = f2bf(ap[j]); }
    v16bf bf = *(const v16bf*)(Wp + ((size_t)(ks * ntiles + nt) * 32 + lane) * 16);
    acc = __builtin_amdgcn_wmma_f32_16x16x32_bf16(false, af, false, bf, (short)0, acc, false, false);
  }
  int col = nt * 16 + (lane & 15);
  int r0  = mt * 16 + ((lane < 16) ? 0 : 8);
  float b  = bias  ? bias[col]  : 0.f;
  float sc = (scale ? scale[col] : 1.f) * scale_c;
  float sh = shift ? shift[col] : 0.f;
#pragma unroll
  for (int v = 0; v < 8; ++v) {
    float x = (acc[v] + b) * sc + sh;
    if (act == 1)      x = fmaxf(x, 0.f);
    else if (act == 2) x = (x >= 0.f) ? x : 0.01f * x;
    Out[(size_t)(r0 + v) * N + col] = x;
  }
}

// Flash attention, D=128 H=4 dh=32 Nseq=512, no masks.
// One wave per (b, h, 16-query tile). Key blocks of 32.
__launch_bounds__(256)
__global__ void mha_flash_kernel(const float* __restrict__ Q, const float* __restrict__ K,
                                 const float* __restrict__ V, float* __restrict__ O) {
  __shared__ __bf16 plds[8][512];
  int lw   = threadIdx.x >> 5;
  int lane = threadIdx.x & 31;
  int gw   = blockIdx.x * 8 + lw;
  int qt = gw & 31;
  int h  = (gw >> 5) & 3;
  int b  = gw >> 7;
  int half = lane >> 4;
  int l16  = lane & 15;
  int kb8  = half * 8;
  int kb16 = half * 16;
  const float qscale = 0.17677669529663687f;   // 1/sqrt(32)
  v16bf qf;
  {
    const float* qp = Q + (size_t)(b * 512 + qt * 16 + l16) * 128 + h * 32 + kb16;
#pragma unroll
    for (int j = 0; j < 16; ++j) { qf[j] = f2bf(qp[j] * qscale); }
  }
  float m[8], l[8];
#pragma unroll
  for (int v = 0; v < 8; ++v) { m[v] = -1e30f; l[v] = 0.f; }
  v8f o0 = {}, o1 = {};
  for (int kt = 0; kt < 16; ++kt) {
    int key0 = kt * 32;
    v16bf k0f, k1f;
    {
      const float* kp = K + (size_t)(b * 512 + key0 + l16) * 128 + h * 32 + kb16;
      __builtin_prefetch(kp + 32 * 128, 0, 1);
#pragma unroll
      for (int j = 0; j < 16; ++j) k0f[j] = f2bf(kp[j]);
      kp += (size_t)16 * 128;
#pragma unroll
      for (int j = 0; j < 16; ++j) k1f[j] = f2bf(kp[j]);
    }
    v8f z = {};
    v8f sA = __builtin_amdgcn_wmma_f32_16x16x32_bf16(false, qf, false, k0f, (short)0, z, false, false);
    v8f sB = __builtin_amdgcn_wmma_f32_16x16x32_bf16(false, qf, false, k1f, (short)0, z, false, false);
#pragma unroll
    for (int v = 0; v < 8; ++v) {
      float t  = half16_max(fmaxf(sA[v], sB[v]));
      float mn = fmaxf(m[v], t);
      float sc = __expf(m[v] - mn);
      float pa = __expf(sA[v] - mn);
      float pb = __expf(sB[v] - mn);
      float rs = half16_sum(pa + pb);
      l[v] = l[v] * sc + rs;
      m[v] = mn;
      o0[v] *= sc; o1[v] *= sc;
      int r = v + 8 * half;
      plds[lw][r * 32 + l16]      = f2bf(pa);
      plds[lw][r * 32 + 16 + l16] = f2bf(pb);
    }
    __syncthreads();
    v16bf pf;
    {
      const __bf16* pp = &plds[lw][l16 * 32 + kb16];
#pragma unroll
      for (int j = 0; j < 16; ++j) { pf[j] = pp[j]; }
    }
    __syncthreads();
    v16bf v0f, v1f;
    {
      const float* vp = V + (size_t)(b * 512 + key0 + kb16) * 128 + h * 32 + l16;
#pragma unroll
      for (int e = 0; e < 16; ++e) {
        v0f[e] = f2bf(vp[(size_t)e * 128]);
        v1f[e] = f2bf(vp[(size_t)e * 128 + 16]);
      }
    }
    o0 = __builtin_amdgcn_wmma_f32_16x16x32_bf16(false, pf, false, v0f, (short)0, o0, false, false);
    o1 = __builtin_amdgcn_wmma_f32_16x16x32_bf16(false, pf, false, v1f, (short)0, o1, false, false);
  }
#pragma unroll
  for (int v = 0; v < 8; ++v) {
    float inv = 1.0f / l[v];
    size_t orow = (size_t)(b * 512 + qt * 16 + v + 8 * half);
    O[orow * 128 + h * 32 + l16]      = o0[v] * inv;
    O[orow * 128 + h * 32 + 16 + l16] = o1[v] * inv;
  }
}

// Out[row, :] = LN(A[row,:] + B[row,:]) * g + beta  (one wave per row)
__global__ void add_ln_kernel(const float* __restrict__ A, const float* __restrict__ Bv,
                              const float* __restrict__ g, const float* __restrict__ be,
                              float* __restrict__ Out, int M, int W, int ostride, int ooff) {
  int wave = (blockIdx.x * blockDim.x + threadIdx.x) >> 5;
  int lane = threadIdx.x & 31;
  if (wave >= M) return;
  int per = W >> 5;
  const float* a = A  + (size_t)wave * W;
  const float* b = Bv + (size_t)wave * W;
  float vals[8];
  float s = 0.f;
  for (int i = 0; i < per; ++i) { int c = lane + (i << 5); float x = a[c] + b[c]; vals[i] = x; s += x; }
  s = wave_sum(s);
  float mu = s / (float)W;
  float vs = 0.f;
  for (int i = 0; i < per; ++i) { float d = vals[i] - mu; vs += d * d; }
  vs = wave_sum(vs);
  float inv = rsqrtf(vs / (float)W + 1e-5f);
  for (int i = 0; i < per; ++i) {
    int c = lane + (i << 5);
    Out[(size_t)wave * ostride + ooff + c] = (vals[i] - mu) * inv * g[c] + be[c];
  }
}

// Out[b, ooff+c] = sum_n In[(b*512+n)*128 + c]
__global__ void seq_sum_kernel(const float* __restrict__ In, float* __restrict__ Out,
                               int ostride, int ooff) {
  int idx = blockIdx.x * blockDim.x + threadIdx.x;
  if (idx >= BB * 128) return;
  int b = idx >> 7, c = idx & 127;
  float s = 0.f;
  const float* p = In + (size_t)b * 512 * 128 + c;
  for (int n = 0; n < 512; ++n) s += p[(size_t)n * 128];
  Out[b * ostride + ooff + c] = s;
}

// tiny cross-attention over batch dim: seq=32, dg=256, H=4, dh=64. grid=H, block 32x32
__global__ void cross_attn_kernel(const float* __restrict__ Q, const float* __restrict__ K,
                                  const float* __restrict__ V, float* __restrict__ Out) {
  __shared__ float sS[32][33];
  __shared__ float sP[32][33];
  int h = blockIdx.x;
  int q = threadIdx.y, k = threadIdx.x;
  float s = 0.f;
  for (int d = 0; d < 64; ++d) s += Q[q * 256 + h * 64 + d] * K[k * 256 + h * 64 + d];
  sS[q][k] = s * 0.125f;   // 64^-0.5
  __syncthreads();
  float mx = -1e30f;
  for (int j = 0; j < 32; ++j) mx = fmaxf(mx, sS[q][j]);
  float e = __expf(sS[q][k] - mx);
  sP[q][k] = e;
  __syncthreads();
  float sum = 0.f;
  for (int j = 0; j < 32; ++j) sum += sP[q][j];
  float invs = 1.f / sum;
  float o0 = 0.f, o1 = 0.f;
  for (int j = 0; j < 32; ++j) {
    float pj = sP[q][j] * invs;
    o0 += pj * V[j * 256 + h * 64 + k];
    o1 += pj * V[j * 256 + h * 64 + 32 + k];
  }
  Out[q * 256 + h * 64 + k]      = o0;
  Out[q * 256 + h * 64 + 32 + k] = o1;
}

// head: out[b] = (relu(hcat[b]@lw1 + lb1)) @ lw2 + lb2 ; hcat [32,512]
__global__ void head_kernel(const float* __restrict__ H, const float* __restrict__ lw1,
                            const float* __restrict__ lb1, const float* __restrict__ lw2,
                            const float* __restrict__ lb2, float* __restrict__ out) {
  __shared__ float h1[64];
  int b = blockIdx.x, c = threadIdx.x;
  float s = lb1[c];
  for (int k = 0; k < 512; ++k) s += H[b * 512 + k] * lw1[k * 64 + c];
  h1[c] = fmaxf(s, 0.f);
  __syncthreads();
  if (c == 0) {
    float o = lb2[0];
    for (int k = 0; k < 64; ++k) o += h1[k] * lw2[k];
    out[b] = o;
  }
}

// ---------------- host launchers ----------------
static void launch_gemm(hipStream_t s, const float* A, const __bf16* Wp, const float* bias,
                        const float* scale, const float* shift, float scale_c,
                        float* Out, int M, int K, int N, int act) {
  int waves = (M >> 4) * (N >> 4);
  int threads = waves * 32;
  int grid = (threads + 255) / 256;
  gemm_wmma_kernel<<<grid, 256, 0, s>>>(A, Wp, bias, scale, shift, scale_c, Out, M, K, N, act);
}
static void launch_pack(hipStream_t s, const float* W, __bf16* Wp, int K, int N) {
  int total = K * N;
  pack_w_kernel<<<(total + 255) / 256, 256, 0, s>>>(W, Wp, K, N);
}
static void launch_copy(hipStream_t s, const float* src, float* dst, size_t n) {
  copy_f32_kernel<<<(int)((n + 255) / 256), 256, 0, s>>>(src, dst, n);
}
static void launch_add_ln(hipStream_t s, const float* A, const float* B, const float* g,
                          const float* be, float* Out, int M, int W, int ostride, int ooff) {
  int threads = M * 32;
  add_ln_kernel<<<(threads + 255) / 256, 256, 0, s>>>(A, B, g, be, Out, M, W, ostride, ooff);
}

extern "C" void kernel_launch(void* const* d_in, const int* in_sizes, int n_in,
                              void* d_out, int out_size, void* d_ws, size_t ws_size,
                              hipStream_t stream) {
  (void)n_in; (void)out_size; (void)ws_size;
  const float* ch1_x = (const float*)d_in[0];
  const float* ch2_x = (const float*)d_in[1];
  const int* e1 = (const int*)d_in[2];
  const int* e2 = (const int*)d_in[3];
  int E = in_sizes[2] / 2;
  const int *e1s = e1, *e1d = e1 + E;
  const int *e2s = e2, *e2d = e2 + E;
  auto P = [&](int i) { return (const float*)d_in[i]; };

  // param leaves, pytree (alphabetical) order, starting at index 6:
  const float *cb_bk = P(6),  *cb_bo = P(7),  *cb_bq = P(8),  *cb_bv = P(9);
  const float *cb_fb1 = P(10), *cb_fb2 = P(11), *cb_fw1 = P(12), *cb_fw2 = P(13);
  const float *cb_n1b = P(14), *cb_n1g = P(15), *cb_n2b = P(16), *cb_n2g = P(17);
  const float *cb_wk = P(18), *cb_wo = P(19), *cb_wq = P(20), *cb_wv = P(21);
  struct ConvP { const float *b1, *b2, *bnb, *bng, *w1, *w2; };
  auto conv = [&](int base) { return ConvP{P(base), P(base + 1), P(base + 2), P(base + 3), P(base + 4), P(base + 5)}; };
  ConvP eac1 = conv(22), eac2 = conv(28), ebc1 = conv(34), ebc2 = conv(40);
  const float *lb1 = P(46), *lb2 = P(47), *lw1 = P(48), *lw2 = P(49);
  const float *m_bk = P(50), *m_bo = P(51), *m_bq = P(52), *m_bv = P(53);
  const float *m_fb1 = P(54), *m_fb2 = P(55), *m_fw1 = P(56), *m_fw2 = P(57);
  const float *m_n1b = P(58), *m_n1g = P(59), *m_n2b = P(60), *m_n2g = P(61);
  const float *m_wk = P(62), *m_wo = P(63), *m_wq = P(64), *m_wv = P(65);

  float* wsf = (float*)d_ws;
  float* agg  = wsf + OFF_AGG;
  float* hmid = wsf + OFF_HMID;
  float* xA1  = wsf + OFF_XA1;
  float* xA2  = wsf + OFF_XA2;
  float* xB1  = wsf + OFF_XB1;
  float* xB2  = wsf + OFF_XB2;
  float* qb   = wsf + OFF_QB;
  float* kb   = wsf + OFF_KB;
  float* vb   = wsf + OFF_VB;
  float* ao   = wsf + OFF_AO;
  float* pj   = wsf + OFF_PJ;
  float* xl   = wsf + OFF_XL;
  float* ff1  = wsf + OFF_FF1;
  float* ff2  = wsf + OFF_FF2;
  float* mo   = wsf + OFF_MO;
  float* hA   = wsf + OFF_HA;
  float* hB   = wsf + OFF_HB;
  float* cq   = wsf + OFF_CQ;
  float* ck   = wsf + OFF_CK;
  float* cv   = wsf + OFF_CV;
  float* cat  = wsf + OFF_CAT;
  float* cpj  = wsf + OFF_CPJ;
  float* cxl  = wsf + OFF_CXL;
  float* cf1  = wsf + OFF_CF1;
  float* cf2  = wsf + OFF_CF2;
  float* hcat = wsf + OFF_HCAT;
  __bf16* pw = (__bf16*)(wsf + OFF_FLOAT_END);

  const float BN_SC = 0.9999950000374996f;   // 1/sqrt(1+1e-5), eval BatchNorm fold

  // ---- pack all GEMM weights into WMMA fragment order (bf16) ----
  launch_pack(stream, eac1.w1, pw + PW_EA_C1_W1, 64, 128);
  launch_pack(stream, eac1.w2, pw + PW_EA_C1_W2, 128, 128);
  launch_pack(stream, eac2.w1, pw + PW_EA_C2_W1, 128, 128);
  launch_pack(stream, eac2.w2, pw + PW_EA_C2_W2, 128, 128);
  launch_pack(stream, ebc1.w1, pw + PW_EB_C1_W1, 64, 128);
  launch_pack(stream, ebc1.w2, pw + PW_EB_C1_W2, 128, 128);
  launch_pack(stream, ebc2.w1, pw + PW_EB_C2_W1, 128, 128);
  launch_pack(stream, ebc2.w2, pw + PW_EB_C2_W2, 128, 128);
  launch_pack(stream, m_wq, pw + PW_M_WQ, 128, 128);
  launch_pack(stream, m_wk, pw + PW_M_WK, 128, 128);
  launch_pack(stream, m_wv, pw + PW_M_WV, 128, 128);
  launch_pack(stream, m_wo, pw + PW_M_WO, 128, 128);
  launch_pack(stream, m_fw1, pw + PW_M_F1, 128, 256);
  launch_pack(stream, m_fw2, pw + PW_M_F2, 256, 128);
  launch_pack(stream, cb_wq, pw + PW_C_WQ, 256, 256);
  launch_pack(stream, cb_wk, pw + PW_C_WK, 256, 256);
  launch_pack(stream, cb_wv, pw + PW_C_WV, 256, 256);
  launch_pack(stream, cb_wo, pw + PW_C_WO, 256, 256);
  launch_pack(stream, cb_fw1, pw + PW_C_F1, 256, 512);
  launch_pack(stream, cb_fw2, pw + PW_C_F2, 512, 256);

  // ---- GIN conv (leaky(relu(.)) == relu(.)) ----
  auto gin_conv = [&](const float* xin, int width, const int* es, const int* ed,
                      const ConvP& c, const __bf16* w1p, const __bf16* w2p, float* xout) {
    launch_copy(stream, xin, agg, (size_t)BNROWS * width);
    int tot = E * (width / 4);
    edge_scatter_kernel<<<(tot + 255) / 256, 256, 0, stream>>>(es, ed, xin, agg, E, width);
    launch_gemm(stream, agg, w1p, c.b1, c.bng, c.bnb, BN_SC, hmid, BNROWS, width, 128, 1);
    launch_gemm(stream, hmid, w2p, c.b2, nullptr, nullptr, 1.f, xout, BNROWS, 128, 128, 1);
  };
  gin_conv(ch1_x, 64,  e1s, e1d, eac1, pw + PW_EA_C1_W1, pw + PW_EA_C1_W2, xA1);
  gin_conv(xA1,   128, e1s, e1d, eac2, pw + PW_EA_C2_W1, pw + PW_EA_C2_W2, xA2);
  gin_conv(ch2_x, 64,  e2s, e2d, ebc1, pw + PW_EB_C1_W1, pw + PW_EB_C1_W2, xB1);
  gin_conv(xB1,   128, e2s, e2d, ebc2, pw + PW_EB_C2_W1, pw + PW_EB_C2_W2, xB2);

  // ---- MHA block (masks all false): Q from A-channel, K/V from B-channel ----
  auto mha_block = [&](const float* Qin, const float* KVin, float* out) {
    launch_gemm(stream, Qin,  pw + PW_M_WQ, m_bq, nullptr, nullptr, 1.f, qb, BNROWS, 128, 128, 0);
    launch_gemm(stream, KVin, pw + PW_M_WK, m_bk, nullptr, nullptr, 1.f, kb, BNROWS, 128, 128, 0);
    launch_gemm(stream, KVin, pw + PW_M_WV, m_bv, nullptr, nullptr, 1.f, vb, BNROWS, 128, 128, 0);
    mha_flash_kernel<<<512, 256, 0, stream>>>(qb, kb, vb, ao);   // 4096 waves
    launch_gemm(stream, ao, pw + PW_M_WO, m_bo, nullptr, nullptr, 1.f, pj, BNROWS, 128, 128, 0);
    launch_add_ln(stream, Qin, pj, m_n1g, m_n1b, xl, BNROWS, 128, 128, 0);
    launch_gemm(stream, xl,  pw + PW_M_F1, m_fb1, nullptr, nullptr, 1.f, ff1, BNROWS, 128, 256, 2);
    launch_gemm(stream, ff1, pw + PW_M_F2, m_fb2, nullptr, nullptr, 1.f, ff2, BNROWS, 256, 128, 0);
    launch_add_ln(stream, xl, ff2, m_n2g, m_n2b, out, BNROWS, 128, 128, 0);
  };
  mha_block(xA1, xB1, mo);
  seq_sum_kernel<<<(BB * 128 + 255) / 256, 256, 0, stream>>>(mo, hA, 256, 0);
  mha_block(xA2, xB2, mo);
  seq_sum_kernel<<<(BB * 128 + 255) / 256, 256, 0, stream>>>(mo, hA, 256, 128);
  // hB = concat(segment_sum(hB1), segment_sum(hB2)) -- contiguous batches
  seq_sum_kernel<<<(BB * 128 + 255) / 256, 256, 0, stream>>>(xB1, hB, 256, 0);
  seq_sum_kernel<<<(BB * 128 + 255) / 256, 256, 0, stream>>>(xB2, hB, 256, 128);

  // ---- cross-attention blocks (shared params), output into hcat halves ----
  auto cross_block = [&](const float* xq, const float* xkv, int hcat_off) {
    launch_gemm(stream, xq,  pw + PW_C_WQ, cb_bq, nullptr, nullptr, 1.f, cq, 32, 256, 256, 0);
    launch_gemm(stream, xkv, pw + PW_C_WK, cb_bk, nullptr, nullptr, 1.f, ck, 32, 256, 256, 0);
    launch_gemm(stream, xkv, pw + PW_C_WV, cb_bv, nullptr, nullptr, 1.f, cv, 32, 256, 256, 0);
    cross_attn_kernel<<<4, dim3(32, 32), 0, stream>>>(cq, ck, cv, cat);
    launch_gemm(stream, cat, pw + PW_C_WO, cb_bo, nullptr, nullptr, 1.f, cpj, 32, 256, 256, 0);
    launch_add_ln(stream, xq, cpj, cb_n1g, cb_n1b, cxl, 32, 256, 256, 0);
    launch_gemm(stream, cxl, pw + PW_C_F1, cb_fb1, nullptr, nullptr, 1.f, cf1, 32, 256, 512, 2);
    launch_gemm(stream, cf1, pw + PW_C_F2, cb_fb2, nullptr, nullptr, 1.f, cf2, 32, 512, 256, 0);
    launch_add_ln(stream, cxl, cf2, cb_n2g, cb_n2b, hcat, 32, 256, 512, hcat_off);
  };
  cross_block(hA, hB, 0);
  cross_block(hB, hA, 256);

  // ---- head ----
  head_kernel<<<BB, 64, 0, stream>>>(hcat, lw1, lb1, lw2, lb2, (float*)d_out);
}
